// ProteinEncoder_71545565217124
// MI455X (gfx1250) — matrix operand order repaired and checked by reading, not compiled, not run on hardware
//
#include <hip/hip_runtime.h>

typedef __attribute__((ext_vector_type(16))) __bf16 v16bf;
typedef __attribute__((ext_vector_type(8)))  float  v8f;

#define DIM0 1280
#define DIM1 512
#define DIM2 300
#define DIM3 300
#define LDPAD 512   // leading dimension (floats) of intermediate feature buffers

// ---------------------------------------------------------------------------
// fp32 -> bf16, round-to-nearest (integer bias add; drops only tie-to-even).
// Pack two results with a single v_perm_b32: bytes {7,6,3,2} of {hi,lo}
// are the two high halfwords -> packed bf16 pair in one VALU op.
// ---------------------------------------------------------------------------
__device__ __forceinline__ unsigned pk2bf(float a, float b) {
  union { float f; unsigned u; } xa, xb; xa.f = a; xb.f = b;
  return __builtin_amdgcn_perm(xb.u + 0x8000u, xa.u + 0x8000u, 0x07060302u);
}
__device__ __forceinline__ unsigned short f2bf(float f) {
  union { float f; unsigned u; } x; x.f = f;
  return (unsigned short)((x.u + 0x8000u) >> 16);
}

// ---------------------------------------------------------------------------
// Tiled bf16 WMMA GEMM:  H[n][out] = A[n][K] * W[K][out]
// Block = 256 threads = 8 waves; block tile 128x64; wave tile 32x32 (2x2 WMMA).
// fp32 inputs converted to bf16 while staging into LDS (vectorized b128 loads).
// OOB rows/cols are CLAMPED, not guarded: garbage only lands in output
// rows/cols that the epilogue masks off.  Only the W K-tail needs zero-fill.
// ---------------------------------------------------------------------------
#define TM 128
#define TN 64
#define TK 32
#define STR 40   // LDS row stride in halfwords (32 data + 8 pad, 16B-multiple)

__global__ __launch_bounds__(256) void k_gemm(
    const float* __restrict__ A, const float* __restrict__ W,
    float* __restrict__ H, int nRows, int Kpad, int Kreal, int outDim,
    int ldA, int ldH)
{
  __shared__ __align__(16) unsigned short sA[TM * STR];
  __shared__ __align__(16) unsigned short sB[TN * STR];

  const int tid  = threadIdx.x;
  const int wid  = tid >> 5;
  const int lane = tid & 31;
  const int l16  = lane & 15;
  const int hl   = lane >> 4;           // half-wave select
  const int waveRow = wid >> 1;         // 0..3 -> 32-row strip
  const int waveCol = wid & 1;          // 0..1 -> 32-col strip
  const int rowBase = blockIdx.y * TM;
  const int colBase = blockIdx.x * TN;

  v8f acc[2][2] = {};

  for (int kb = 0; kb < Kpad; kb += TK) {
    __syncthreads();
    // Stage A tile: TM x TK fp32 -> bf16, 4 x b128 loads + 4 x b64 LDS stores
#pragma unroll
    for (int j = 0; j < (TM * TK) / (256 * 4); ++j) {
      int idx = j * 256 + tid;
      int r = idx >> 3, k4 = idx & 7;
      int gr = rowBase + r;
      gr = gr < nRows ? gr : nRows - 1;               // clamp, masked at store
      const float4 v = *(const float4*)(A + (size_t)gr * ldA + kb + k4 * 4);
      uint2 p; p.x = pk2bf(v.x, v.y); p.y = pk2bf(v.z, v.w);
      *(uint2*)&sA[r * STR + k4 * 4] = p;
    }
    // Stage W tile transposed: sB[n][k] = W[kb+k][colBase+n]
    // dims are multiples of 4, so each float4 is entirely in- or out-of-range.
#pragma unroll
    for (int j = 0; j < (TN * TK) / (256 * 4); ++j) {
      int idx = j * 256 + tid;
      int n4 = idx & 15, k = idx >> 4;
      int gk = kb + k;
      int gn = colBase + n4 * 4;
      float4 v = make_float4(0.f, 0.f, 0.f, 0.f);
      if (gk < Kreal && gn < outDim)                  // K-tail must be zero
        v = *(const float4*)(W + (size_t)gk * outDim + gn);
      unsigned short* p = &sB[(n4 * 4) * STR + k];
      p[0 * STR] = f2bf(v.x);
      p[1 * STR] = f2bf(v.y);
      p[2 * STR] = f2bf(v.z);
      p[3 * STR] = f2bf(v.w);
    }
    // Prefetch next A K-tile (global_prefetch_b8), clamped address
    if (kb + TK < Kpad) {
      int gr = rowBase + (tid & 127);
      gr = gr < nRows ? gr : nRows - 1;
      __builtin_prefetch(A + (size_t)gr * ldA + (kb + TK), 0, 0);
    }
    __syncthreads();

    // Build fragments per CDNA5 WMMA VGPR layouts (two b128 LDS reads each).
    union { v16bf v; uint4 q[2]; } a[2], b[2];
#pragma unroll
    for (int mi = 0; mi < 2; ++mi) {
      // A 16x32 bf16: lane l16 = row M; V0-3 -> K = hl*8+0..7, V4-7 -> +16
      const unsigned short* p = &sA[(waveRow * 32 + mi * 16 + l16) * STR];
      a[mi].q[0] = *(const uint4*)(p + hl * 8);
      a[mi].q[1] = *(const uint4*)(p + 16 + hl * 8);
    }
#pragma unroll
    for (int ni = 0; ni < 2; ++ni) {
      // B 32x16 bf16: lane l16 = col N; V0-3 -> K = hl*8+0..7, V4-7 -> +16
      const unsigned short* p = &sB[(waveCol * 32 + ni * 16 + l16) * STR];
      b[ni].q[0] = *(const uint4*)(p + hl * 8);
      b[ni].q[1] = *(const uint4*)(p + 16 + hl * 8);
    }
#pragma unroll
    for (int mi = 0; mi < 2; ++mi)
#pragma unroll
      for (int ni = 0; ni < 2; ++ni)
        acc[mi][ni] = __builtin_amdgcn_wmma_f32_16x16x32_bf16(
            false, a[mi].v, false, b[ni].v, (short)0, acc[mi][ni],
            false, false);
  }

  // Store: C/D layout -> lane l16 = col N, row M = r + 8*hl within 16x16 tile
#pragma unroll
  for (int mi = 0; mi < 2; ++mi) {
#pragma unroll
    for (int ni = 0; ni < 2; ++ni) {
      int c = colBase + waveCol * 32 + ni * 16 + l16;
      if (c >= outDim) continue;
#pragma unroll
      for (int r = 0; r < 8; ++r) {
        int m = rowBase + waveRow * 32 + mi * 16 + hl * 8 + r;
        if (m < nRows) H[(size_t)m * ldH + c] = acc[mi][ni][r];
      }
    }
  }
}

// ---------------------------------------------------------------------------
// Degree / normalization kernels
// ---------------------------------------------------------------------------
__global__ void k_deg_init(float* deg, int n) {
  int i = blockIdx.x * blockDim.x + threadIdx.x;
  if (i < n) deg[i] = 1.0f;   // self-loop contributes 1
}
__global__ void k_deg_count(const int* __restrict__ dst, float* deg, int E) {
  int e = blockIdx.x * blockDim.x + threadIdx.x;
  if (e < E) atomicAdd(&deg[dst[e]], 1.0f);
}
__global__ void k_dinv(const float* __restrict__ deg, float* dinv, int n) {
  int i = blockIdx.x * blockDim.x + threadIdx.x;
  if (i < n) {
    float d = deg[i];
    dinv[i] = d > 0.0f ? rsqrtf(d) : 0.0f;
  }
}

// OUT[i] = H[i] * dinv[i]^2   (self-loop term, also initializes OUT). float4.
__global__ void k_selfloop(const float* __restrict__ H, float* __restrict__ OUT,
                           const float* __restrict__ dinv, int outDim4, int ld) {
  int i = blockIdx.x;
  float w = dinv[i] * dinv[i];
  const float4* hi = (const float4*)(H + (size_t)i * ld);
  float4* oi = (float4*)(OUT + (size_t)i * ld);
  for (int c = threadIdx.x; c < outDim4; c += blockDim.x) {
    float4 v = hi[c];
    v.x *= w; v.y *= w; v.z *= w; v.w *= w;
    oi[c] = v;
  }
}

// One wave per edge: OUT[dst] += H[src] * dinv[src]*dinv[dst].  float4 gather,
// per-component f32 atomic scatter (global_atomic_add_f32).
__global__ __launch_bounds__(256) void k_scatter(
    const float* __restrict__ H, float* __restrict__ OUT,
    const int* __restrict__ src, const int* __restrict__ dst,
    const float* __restrict__ dinv, int E, int outDim4, int ld)
{
  int w = (blockIdx.x * blockDim.x + threadIdx.x) >> 5;
  int lane = threadIdx.x & 31;
  if (w >= E) return;
  int s = src[w], d = dst[w];
  float nrm = dinv[s] * dinv[d];
  const float4* hs = (const float4*)(H + (size_t)s * ld);
  float* od = OUT + (size_t)d * ld;
  for (int c = lane; c < outDim4; c += 32) {
    float4 h = hs[c];
    atomicAdd(&od[c * 4 + 0], h.x * nrm);
    atomicAdd(&od[c * 4 + 1], h.y * nrm);
    atomicAdd(&od[c * 4 + 2], h.z * nrm);
    atomicAdd(&od[c * 4 + 3], h.w * nrm);
  }
}

// OUT[i][c] = relu(IN[i][c] + bias[c]) for c<outDim; 0 for outDim<=c<padDim.
// All dims multiples of 4 -> float4 path; each float4 is entirely in/out.
__global__ void k_finalize(const float* __restrict__ IN, float* __restrict__ OUT,
                           const float* __restrict__ bias,
                           int outDim4, int padDim4, int ldIn, int ldOut) {
  int i = blockIdx.x;
  const float4* ii = (const float4*)(IN + (size_t)i * ldIn);
  const float4* bb = (const float4*)bias;
  float4* oi = (float4*)(OUT + (size_t)i * ldOut);
  for (int c = threadIdx.x; c < padDim4; c += blockDim.x) {
    float4 v = make_float4(0.f, 0.f, 0.f, 0.f);
    if (c < outDim4) {
      float4 a = ii[c], b = bb[c];
      v.x = fmaxf(a.x + b.x, 0.f);
      v.y = fmaxf(a.y + b.y, 0.f);
      v.z = fmaxf(a.z + b.z, 0.f);
      v.w = fmaxf(a.w + b.w, 0.f);
    }
    oi[c] = v;
  }
}

// ---------------------------------------------------------------------------
extern "C" void kernel_launch(void* const* d_in, const int* in_sizes, int n_in,
                              void* d_out, int out_size, void* d_ws, size_t ws_size,
                              hipStream_t stream) {
  const float* x  = (const float*)d_in[0];
  const float* W1 = (const float*)d_in[1];
  const float* b1 = (const float*)d_in[2];
  const float* W2 = (const float*)d_in[3];
  const float* b2 = (const float*)d_in[4];
  const float* W3 = (const float*)d_in[5];
  const float* b3 = (const float*)d_in[6];
  const int*   ei = (const int*)d_in[7];

  const int n = in_sizes[0] / DIM0;
  const int E = in_sizes[7] / 2;
  const int* src = ei;
  const int* dst = ei + E;
  float* out = (float*)d_out;

  char* ws = (char*)d_ws;
  const size_t NB = (size_t)n * LDPAD * sizeof(float);
  float* B0   = (float*)ws;              // GEMM output (H)
  float* B1   = (float*)(ws + NB);       // aggregation output / next GEMM input
  float* dinv = (float*)(ws + 2 * NB);
  float* deg  = dinv + n;

  dim3 blk(256);

  // degree -> D^-1/2
  k_deg_init <<<(n + 255) / 256, blk, 0, stream>>>(deg, n);
  k_deg_count<<<(E + 255) / 256, blk, 0, stream>>>(dst, deg, E);
  k_dinv     <<<(n + 255) / 256, blk, 0, stream>>>(deg, dinv, n);

  const int scatBlocks = (E + 7) / 8;    // 8 waves per 256-thread block

  // ---- Layer 1: x[1280] @ W1 -> 512 ----
  {
    dim3 grid((DIM1 + TN - 1) / TN, (n + TM - 1) / TM);
    k_gemm<<<grid, blk, 0, stream>>>(x, W1, B0, n, DIM0, DIM0, DIM1, DIM0, LDPAD);
    k_selfloop<<<n, blk, 0, stream>>>(B0, B1, dinv, DIM1 / 4, LDPAD);
    k_scatter <<<scatBlocks, blk, 0, stream>>>(B0, B1, src, dst, dinv, E, DIM1 / 4, LDPAD);
    k_finalize<<<n, blk, 0, stream>>>(B1, B1, b1, DIM1 / 4, DIM1 / 4, LDPAD, LDPAD);
  }
  // ---- Layer 2: B1[512] @ W2 -> 300 (zero-pad K for next layer to 320) ----
  {
    dim3 grid((DIM2 + TN - 1) / TN, (n + TM - 1) / TM);
    k_gemm<<<grid, blk, 0, stream>>>(B1, W2, B0, n, DIM1, DIM1, DIM2, LDPAD, LDPAD);
    k_selfloop<<<n, blk, 0, stream>>>(B0, B1, dinv, DIM2 / 4, LDPAD);
    k_scatter <<<scatBlocks, blk, 0, stream>>>(B0, B1, src, dst, dinv, E, DIM2 / 4, LDPAD);
    k_finalize<<<n, blk, 0, stream>>>(B1, B1, b2, DIM2 / 4, 320 / 4, LDPAD, LDPAD);
  }
  // ---- Layer 3: B1[300 padded to 320] @ W3 -> 300, dense output ----
  {
    dim3 grid((DIM3 + TN - 1) / TN, (n + TM - 1) / TM);
    k_gemm<<<grid, blk, 0, stream>>>(B1, W3, B0, n, 320, DIM2, DIM3, LDPAD, LDPAD);
    k_selfloop<<<n, blk, 0, stream>>>(B0, B1, dinv, DIM3 / 4, LDPAD);
    k_scatter <<<scatBlocks, blk, 0, stream>>>(B0, B1, src, dst, dinv, E, DIM3 / 4, LDPAD);
    k_finalize<<<n, blk, 0, stream>>>(B1, out, b3, DIM3 / 4, DIM3 / 4, LDPAD, DIM3);
  }
}
